// SpatioTemporalMLPAttention_59047210385825
// MI455X (gfx1250) — matrix operand order repaired and checked by reading, not compiled, not run on hardware
//
#include <hip/hip_runtime.h>
#include <hip/hip_bf16.h>

typedef __attribute__((ext_vector_type(16))) _Float16 v16h;
typedef __attribute__((ext_vector_type(8)))  _Float16 v8h;
typedef __attribute__((ext_vector_type(8)))  float    v8f;

#define B_   4
#define Q_   384
#define K_   384
#define F_   64
#define DVN_ 32
#define FV_  64
#define H1_  256
#define EPS_ 1e-6f

// gelu (tanh approximation), branch-free:
//   gelu(x) = 0.5x(1+tanh(c(x+0.044715x^3))) = x - x * rcp(exp(2c(x+0.044715x^3)) + 1)
// exp->inf  => gelu -> x ; exp->0 => gelu -> 0  (correct saturation, no branches)
__device__ __forceinline__ float gelu_f(float x) {
    float u = 1.5957691216057308f * (x + 0.044715f * x * x * x);   // 2*sqrt(2/pi)
    float e = __expf(u);                                           // v_exp_f32 path
    return x - x * __builtin_amdgcn_rcpf(e + 1.0f);                // v_rcp_f32
}

// ---------------------------------------------------------------------------
// Prep: P[b,q,256] and R[b,k,256] (separable part of attn layer 1)
// ---------------------------------------------------------------------------
__global__ void prep_pr(const float* __restrict__ qs, const float* __restrict__ ks,
                        const float* __restrict__ qs_s, const float* __restrict__ ks_s,
                        const float* __restrict__ qs_t, const float* __restrict__ ks_t,
                        const float* __restrict__ vnode,
                        const float* __restrict__ W1, const float* __restrict__ b1,
                        float* __restrict__ P, float* __restrict__ R)
{
    const int n = blockIdx.x;
    const int c = threadIdx.x;            // 0..255
    if (n < B_ * Q_) {
        const float* q = qs + n * F_;
        float acc = 0.0f;
        #pragma unroll 8
        for (int f = 0; f < F_; ++f) acc += q[f] * W1[f * H1_ + c];
        acc += qs_s[n * 2 + 0] * W1[160 * H1_ + c] + qs_s[n * 2 + 1] * W1[161 * H1_ + c];
        acc += qs_t[n] * W1[164 * H1_ + c];
        P[n * H1_ + c] = acc;
    } else {
        const int m = n - B_ * Q_;
        const int b = m / K_;
        const float* kr = ks + m * F_;
        float acc = b1[c];
        #pragma unroll 8
        for (int f = 0; f < F_; ++f) acc += kr[f] * W1[(64 + f) * H1_ + c];
        #pragma unroll 8
        for (int j = 0; j < DVN_; ++j) acc += vnode[b * DVN_ + j] * W1[(128 + j) * H1_ + c];
        acc -= ks_s[m * 2 + 0] * W1[160 * H1_ + c] + ks_s[m * 2 + 1] * W1[161 * H1_ + c];
        acc -= ks_t[m] * W1[164 * H1_ + c];
        R[m * H1_ + c] = acc;
    }
}

// ---------------------------------------------------------------------------
// Prep: vproj = vs @ Wv + bv   (B*K rows, 64 wide)
// ---------------------------------------------------------------------------
__global__ void prep_vproj(const float* __restrict__ vs, const float* __restrict__ Wv,
                           const float* __restrict__ bv, float* __restrict__ vproj)
{
    const int n = blockIdx.x;
    const int d = threadIdx.x;            // 0..63
    const float* r = vs + n * FV_;
    float acc = bv[d];
    #pragma unroll 8
    for (int f = 0; f < FV_; ++f) acc += r[f] * Wv[f * 64 + d];
    vproj[n * 64 + d] = acc;
}

// ---------------------------------------------------------------------------
// Prep: gate = (gelu(vs @ Wg1 + bg1)) @ Wg2 + bg2
// ---------------------------------------------------------------------------
__global__ void prep_gate(const float* __restrict__ vs,
                          const float* __restrict__ Wg1, const float* __restrict__ bg1,
                          const float* __restrict__ Wg2, const float* __restrict__ bg2,
                          float* __restrict__ gate)
{
    __shared__ float sh[256];
    const int n = blockIdx.x;
    const int c = threadIdx.x;            // 0..255
    const float* r = vs + n * FV_;
    float acc = bg1[c];
    #pragma unroll 8
    for (int f = 0; f < FV_; ++f) acc += r[f] * Wg1[f * 256 + c];
    sh[c] = gelu_f(acc);
    __syncthreads();
    if (c < 64) {
        float g = bg2[c];
        for (int h = 0; h < 256; ++h) g += sh[h] * Wg2[h * 64 + c];
        gate[n * 64 + c] = g;
    }
}

// ---------------------------------------------------------------------------
// Prep: repack W2 (256x256 f32) into f16 WMMA B-fragment order.
// Bpack[((kstep*16+ct)*32+lane)*16 + h] = W2[kstep*32 + (lane<16 ? h : 16+h)][ct*16 + lane%16]
// ---------------------------------------------------------------------------
__global__ void prep_pack(const float* __restrict__ W2, _Float16* __restrict__ Bpack)
{
    const int idx   = blockIdx.x * 256 + threadIdx.x;   // 0..65535
    const int h     = idx & 15;
    const int lane  = (idx >> 4) & 31;
    const int ctks  = idx >> 9;                         // 0..127
    const int ct    = ctks & 15;
    const int kstep = ctks >> 4;
    const int krow  = kstep * 32 + ((lane < 16) ? h : 16 + h);
    const int col   = ct * 16 + (lane & 15);
    Bpack[idx] = (_Float16)W2[krow * 256 + col];
}

__global__ void init_acc(float* __restrict__ vacc) { vacc[threadIdx.x] = 0.0f; }

// ---------------------------------------------------------------------------
// Main: one block per (b,q). 256 threads = 8 wave32.
// ---------------------------------------------------------------------------
__global__ void __launch_bounds__(256, 1)
attn_main(const float* __restrict__ qs_s, const float* __restrict__ ks_s,
          const float* __restrict__ qs_t, const float* __restrict__ ks_t,
          const int*   __restrict__ valid_lens,
          const float* __restrict__ W1,  const float* __restrict__ b2v,
          const float* __restrict__ W3,  const float* __restrict__ b3v,
          const float* __restrict__ ln1s, const float* __restrict__ ln1b,
          const float* __restrict__ P,   const float* __restrict__ R,
          const float* __restrict__ vproj, const float* __restrict__ gate,
          const _Float16* __restrict__ Bpack,
          float* __restrict__ vacc, float* __restrict__ out_ctx)
{
    const int q    = blockIdx.x;
    const int b    = blockIdx.y;
    const int tid  = threadIdx.x;
    const int lane = tid & 31;
    const int wave = tid >> 5;

    __shared__ float sP[H1_];
    __shared__ __align__(16) _Float16 sA[16 * H1_];   // gelu(h1) row tile, f16
    __shared__ float sScore[K_];
    __shared__ int   sMask[K_];
    __shared__ float sW162[H1_], sW163[H1_];
    __shared__ float sB2[H1_],  sW3[H1_];
    __shared__ float sCtxP[256];
    __shared__ float sRed[2];

    const int nq = b * Q_ + q;
    sP[tid]    = P[nq * H1_ + tid];
    sW162[tid] = W1[162 * H1_ + tid];
    sW163[tid] = W1[163 * H1_ + tid];
    sB2[tid]   = b2v[tid];
    sW3[tid]   = W3[tid];
    const float qs0 = qs_s[nq * 2 + 0], qs1 = qs_s[nq * 2 + 1];
    const float qt  = qs_t[nq];
    const int   vl  = valid_lens[b];
    __syncthreads();

    const int kk   = tid >> 4;          // 0..15: row within tile
    const int cseg = (tid & 15) * 16;   // 16-col segment

    for (int rt = 0; rt < K_ / 16; ++rt) {
        const int k0 = rt * 16;
        if (tid < 16) sScore[k0 + tid] = 0.0f;
        // ---- build f16 A tile: gelu(P + R + sd*w162 + sd2*w163) ----
        {
            const int k  = k0 + kk;
            const int nk = b * K_ + k;
            const float d0  = qs0 - ks_s[nk * 2 + 0];
            const float d1  = qs1 - ks_s[nk * 2 + 1];
            const float sd2 = d0 * d0 + d1 * d1;
            const float sd  = sqrtf(sd2);
            if ((tid & 15) == 0)
                sMask[k] = (k < vl) && ((qt - ks_t[nk]) > 0.0f);
            const float* Rrow = R + (size_t)nk * H1_;
            #pragma unroll
            for (int j = 0; j < 16; ++j) {
                const int c = cseg + j;
                float h = sP[c] + Rrow[c] + sd * sW162[c] + sd2 * sW163[c];
                sA[kk * H1_ + c] = (_Float16)gelu_f(h);
            }
        }
        __syncthreads();

        // ---- layer 2 via WMMA: wave owns col-tiles {2w, 2w+1} ----
        float sacc[8];
        #pragma unroll
        for (int j = 0; j < 8; ++j) sacc[j] = 0.0f;
        const int m    = lane & 15;
        const int off0 = (lane < 16) ? 0 : 16;

        #pragma unroll
        for (int cti = 0; cti < 2; ++cti) {
            const int ct = wave * 2 + cti;
            v8f acc = {};
            #pragma unroll
            for (int kstep = 0; kstep < 8; ++kstep) {
                const _Float16* ap = &sA[m * H1_ + kstep * 32 + off0];
                v8h alo = *(const v8h*)(ap);
                v8h ahi = *(const v8h*)(ap + 8);
                v16h a  = __builtin_shufflevector(alo, ahi,
                           0,1,2,3,4,5,6,7,8,9,10,11,12,13,14,15);
                const _Float16* bp = Bpack + (((kstep * 16 + ct) * 32 + lane) << 4);
                v8h blo = *(const v8h*)(bp);
                v8h bhi = *(const v8h*)(bp + 8);
                v16h bb = __builtin_shufflevector(blo, bhi,
                           0,1,2,3,4,5,6,7,8,9,10,11,12,13,14,15);
                acc = __builtin_amdgcn_wmma_f32_16x16x32_f16(
                          false, a, false, bb, (short)0, acc, false, false);
            }
            const int   col = ct * 16 + m;
            const float bc  = sB2[col];
            const float wc  = sW3[col];
            #pragma unroll
            for (int j = 0; j < 8; ++j) sacc[j] += gelu_f(acc[j] + bc) * wc;
        }
        // reduce over the 16-lane half (cols) -> per-row partial score
        #pragma unroll
        for (int sh = 1; sh < 16; sh <<= 1) {
            #pragma unroll
            for (int j = 0; j < 8; ++j) sacc[j] += __shfl_xor(sacc[j], sh, 32);
        }
        if (m == 0) {
            const int rbase = k0 + ((lane >> 4) << 3);
            #pragma unroll
            for (int j = 0; j < 8; ++j) atomicAdd(&sScore[rbase + j], sacc[j]);
        }
        __syncthreads();
    }

    // ---- ctx = masked(attn) @ vproj ----
    const float b3 = b3v[0];
    {
        const int d = tid & 63;
        const int g = tid >> 6;                 // 0..3
        float p = 0.0f;
        for (int k = g * 96; k < g * 96 + 96; ++k) {
            const float a = sMask[k] ? (sScore[k] + b3) : 0.0f;
            p += a * vproj[(size_t)(b * K_ + k) * 64 + d];
        }
        sCtxP[tid] = p;
    }
    __syncthreads();
    float ctxv = 0.0f;
    if (tid < 64) {
        ctxv = sCtxP[tid] + sCtxP[64 + tid] + sCtxP[128 + tid] + sCtxP[192 + tid];
        sCtxP[tid] = ctxv;
    }
    __syncthreads();
    if (tid == 0) {
        float mu = 0.0f;
        for (int i = 0; i < 64; ++i) mu += sCtxP[i];
        mu *= (1.0f / 64.0f);
        float var = 0.0f;
        for (int i = 0; i < 64; ++i) { float d = sCtxP[i] - mu; var += d * d; }
        var *= (1.0f / 64.0f);
        sRed[0] = mu; sRed[1] = rsqrtf(var + EPS_);
    }
    __syncthreads();
    if (tid < 64) {
        const float o = (ctxv - sRed[0]) * sRed[1] * ln1s[tid] + ln1b[tid];
        out_ctx[(size_t)nq * 64 + tid] = o;
        if (q < vl)   // wm[b,q]: contribute to masked mean over axis 1
            atomicAdd(&vacc[b * 64 + tid], gate[(size_t)(b * K_ + q) * 64 + tid] * o);
    }
}

// ---------------------------------------------------------------------------
// Finalize vnode_out = layernorm(vacc / valid_len)
// ---------------------------------------------------------------------------
__global__ void finalize_vn(const float* __restrict__ vacc, const int* __restrict__ valid_lens,
                            const float* __restrict__ ln2s, const float* __restrict__ ln2b,
                            float* __restrict__ out_vn)
{
    __shared__ float sv[4][64];
    __shared__ float st[4][2];
    const int tid = threadIdx.x;       // 256
    const int b = tid >> 6, d = tid & 63;
    const float v = vacc[b * 64 + d] / (float)valid_lens[b];
    sv[b][d] = v;
    __syncthreads();
    if (d == 0) {
        float mu = 0.0f;
        for (int i = 0; i < 64; ++i) mu += sv[b][i];
        mu *= (1.0f / 64.0f);
        float var = 0.0f;
        for (int i = 0; i < 64; ++i) { float x = sv[b][i] - mu; var += x * x; }
        var *= (1.0f / 64.0f);
        st[b][0] = mu; st[b][1] = rsqrtf(var + EPS_);
    }
    __syncthreads();
    out_vn[b * 64 + d] = (v - st[b][0]) * st[b][1] * ln2s[d] + ln2b[d];
}

// ---------------------------------------------------------------------------
extern "C" void kernel_launch(void* const* d_in, const int* in_sizes, int n_in,
                              void* d_out, int out_size, void* d_ws, size_t ws_size,
                              hipStream_t stream)
{
    (void)in_sizes; (void)n_in; (void)out_size; (void)ws_size;
    const float* qs    = (const float*)d_in[0];
    const float* ks    = (const float*)d_in[1];
    const float* vs    = (const float*)d_in[2];
    const float* qs_s  = (const float*)d_in[3];
    const float* ks_s  = (const float*)d_in[4];
    const float* qs_t  = (const float*)d_in[5];
    const float* ks_t  = (const float*)d_in[6];
    const float* vnode = (const float*)d_in[7];
    const int*   vlen  = (const int*)d_in[8];
    const float* Wv    = (const float*)d_in[9];
    const float* bv    = (const float*)d_in[10];
    const float* W1    = (const float*)d_in[11];
    const float* b1    = (const float*)d_in[12];
    const float* W2    = (const float*)d_in[13];
    const float* b2    = (const float*)d_in[14];
    const float* W3    = (const float*)d_in[15];
    const float* b3    = (const float*)d_in[16];
    const float* Wg1   = (const float*)d_in[17];
    const float* bg1   = (const float*)d_in[18];
    const float* Wg2   = (const float*)d_in[19];
    const float* bg2   = (const float*)d_in[20];
    const float* ln1s  = (const float*)d_in[21];
    const float* ln1b  = (const float*)d_in[22];
    const float* ln2s  = (const float*)d_in[23];
    const float* ln2b  = (const float*)d_in[24];

    float* ws    = (float*)d_ws;
    float* P     = ws;                    // B*Q*256 = 393216
    float* R     = ws + 393216;           // B*K*256 = 393216
    float* vproj = ws + 786432;           // B*K*64  = 98304
    float* gate  = ws + 884736;           // B*K*64  = 98304
    float* vacc  = ws + 983040;           // B*64    = 256
    _Float16* Bpack = (_Float16*)(ws + 983296);   // 65536 halves

    float* out_ctx = (float*)d_out;
    float* out_vn  = out_ctx + (size_t)B_ * Q_ * 64;

    init_acc<<<1, 256, 0, stream>>>(vacc);
    prep_pr<<<B_ * Q_ + B_ * K_, 256, 0, stream>>>(qs, ks, qs_s, ks_s, qs_t, ks_t,
                                                   vnode, W1, b1, P, R);
    prep_vproj<<<B_ * K_, 64, 0, stream>>>(vs, Wv, bv, vproj);
    prep_gate<<<B_ * K_, 256, 0, stream>>>(vs, Wg1, bg1, Wg2, bg2, gate);
    prep_pack<<<256, 256, 0, stream>>>(W2, Bpack);

    dim3 grid(Q_, B_);
    attn_main<<<grid, 256, 0, stream>>>(qs_s, ks_s, qs_t, ks_t, vlen,
                                        W1, b2, W3, b3, ln1s, ln1b,
                                        P, R, vproj, gate, Bpack, vacc, out_ctx);
    finalize_vn<<<1, 256, 0, stream>>>(vacc, vlen, ln2s, ln2b, out_vn);
}
